// lstm_model_attention_25752623906937
// MI455X (gfx1250) — matrix operand order repaired and checked by reading, not compiled, not run on hardware
//
#include <hip/hip_runtime.h>
#include <hip/hip_bf16.h>

typedef __attribute__((ext_vector_type(16))) _Float16 v16h;
typedef __attribute__((ext_vector_type(8)))  _Float16 v8h;
typedef __attribute__((ext_vector_type(8)))  float    v8f;

constexpr int kSeq   = 1024;
constexpr int kBatch = 32;
constexpr int kIn    = 256;
constexpr int kHid   = 512;
constexpr int kKTot  = kIn + kHid;     // 768 combined K (x·w then h·u)
constexpr int kNChunk = kKTot / 32;    // 24 K-chunks of 32
constexpr int kXChunk = kIn / 32;      // 8 chunks from x·w
constexpr int kHChunk = kHid / 32;     // 16 chunks from h·u
constexpr int kNWG   = 32;             // one 16-col hidden slice per WG
constexpr int kThreads = 64;           // 2 wave32 per WG (m-tiles 0..15 / 16..31)
constexpr int kLdsBytes = 4 * kNChunk * 32 * 16 * 2;  // 96 KB fragment cache

__device__ __forceinline__ float sigmoidf_(float x) {
  return 1.0f / (1.0f + __expf(-x));
}
// branch-free tanh: 2*sigmoid(2x) - 1  (pure v_exp/v_rcp, no libm divergence)
__device__ __forceinline__ float tanhf_(float x) {
  return 2.0f / (1.0f + __expf(-2.0f * x)) - 1.0f;
}

__device__ __forceinline__ v16h load_frag_g(const _Float16* p, int half8) {
  v8h lo = *(const v8h*)(p + half8);
  v8h hi = *(const v8h*)(p + 16 + half8);
  return __builtin_shufflevector(lo, hi, 0, 1, 2, 3, 4, 5, 6, 7,
                                 8, 9, 10, 11, 12, 13, 14, 15);
}
__device__ __forceinline__ v16h load_frag_lds(const _Float16* p) {
  v8h lo = *(const v8h*)p;
  v8h hi = *(const v8h*)(p + 8);
  return __builtin_shufflevector(lo, hi, 0, 1, 2, 3, 4, 5, 6, 7,
                                 8, 9, 10, 11, 12, 13, 14, 15);
}

// ---------------- prep: x fp32 -> f16 (same [b][t][k] layout) ----------------
__global__ void pack_x_kernel(const float* __restrict__ x,
                              _Float16* __restrict__ x16, int n) {
  int i = blockIdx.x * blockDim.x + threadIdx.x;
  if (i < n) x16[i] = (_Float16)x[i];
}

// ---------------- prep: zero h double-buffer + barrier counter ---------------
__global__ void init_state_kernel(_Float16* __restrict__ h16, int* __restrict__ bar) {
  int i = blockIdx.x * blockDim.x + threadIdx.x;
  if (i < 2 * kBatch * kHid) h16[i] = (_Float16)0.0f;
  if (i == 0) *bar = 0;
}

// ---------------- persistent LSTM recurrence ---------------------------------
__launch_bounds__(kThreads)
__global__ void lstm_persistent_kernel(
    const float* __restrict__ w_i, const float* __restrict__ u_i, const float* __restrict__ b_i,
    const float* __restrict__ w_f, const float* __restrict__ u_f, const float* __restrict__ b_f,
    const float* __restrict__ w_c, const float* __restrict__ u_c, const float* __restrict__ b_c,
    const float* __restrict__ w_o, const float* __restrict__ u_o, const float* __restrict__ b_o,
    const _Float16* __restrict__ x16,   // [32][1024][256] f16
    _Float16* __restrict__ h16,         // [2][32][512] f16 double buffer
    int* __restrict__ bar,
    float* __restrict__ out)            // hidden_seq [1024][32][512] ++ h_t ++ c_t
{
  extern __shared__ _Float16 ldsB[];    // [gate][chunk][lane][16] f16 fragments

  const int tid  = threadIdx.x;
  const int lane = tid & 31;
  const int wid  = tid >> 5;            // m-tile (0 or 1)
  const int n0   = blockIdx.x * 16;     // hidden column base of this WG
  const int m0   = wid * 16;            // batch row base of this wave

  const float* Wg[4] = {w_i, w_f, w_c, w_o};
  const float* Ug[4] = {u_i, u_f, u_c, u_o};
  const float* Bg[4] = {b_i, b_f, b_c, b_o};

  // ---- one-time: pack B fragments (w stacked over u, f16, fragment order) ---
  // B layout per ISA: 32x16 tile; lanes 0-15 hold K=0..7,16..23 (N=lane),
  // lanes 16-31 hold K=8..15,24..31 (N=lane-16); matches A fragment order.
  for (int e = tid; e < 4 * kNChunk * 512; e += kThreads) {
    int elem = e & 15;                  // K position within lane (0..15)
    int ln   = (e >> 4) & 31;           // target lane
    int gk   = e >> 9;
    int kc   = gk % kNChunk;
    int g    = gk / kNChunk;
    int n    = n0 + (ln & 15);
    int kl   = ((ln >> 4) & 1) * 8 + (elem & 7) + (elem >> 3) * 16;
    int k    = kc * 32 + kl;
    float v  = (k < kIn) ? Wg[g][k * kHid + n] : Ug[g][(k - kIn) * kHid + n];
    ldsB[e]  = (_Float16)v;
  }
  __syncthreads();

  // per-lane biases (each lane owns output column n0 + (lane&15));
  // added in the elementwise tail so accumulators start from inline C=0.
  float bias[4];
#pragma unroll
  for (int g = 0; g < 4; ++g) bias[g] = Bg[g][n0 + (lane & 15)];

  // cell state lives in registers for the whole sequence (tile is wave-owned)
  float creg[8];
#pragma unroll
  for (int r = 0; r < 8; ++r) creg[r] = 0.0f;

  const int    half8 = ((lane >> 4) & 1) * 8;    // A-frag K sub-offset per ISA
  const int    arow  = m0 + (lane & 15);         // A-matrix row (batch) for lane
  const size_t xrow  = (size_t)arow * kSeq * kIn;
  const int    jcol  = n0 + (lane & 15);         // C/D column for lane

  for (int t = 0; t < kSeq; ++t) {
    const _Float16* __restrict__ hprev = h16 + (size_t)(t & 1) * (kBatch * kHid);
    _Float16* __restrict__ hnext       = h16 + (size_t)((t + 1) & 1) * (kBatch * kHid);

    v8f acc[4] = {};                    // C starts at inline 0; bias added in tail

    const _Float16* xa = x16 + xrow + (size_t)t * kIn;
    if (t + 1 < kSeq) __builtin_prefetch(xa + kIn, 0, 1);   // global_prefetch_b8

    // ---- K-phase 1: input projection (x tile, chunks 0..7) ----
#pragma unroll
    for (int kc = 0; kc < kXChunk; ++kc) {
      v16h a = load_frag_g(xa + kc * 32, half8);
      v16h bfr[4];
#pragma unroll
      for (int g = 0; g < 4; ++g)       // issue all 8 ds_load_b128 first
        bfr[g] = load_frag_lds(ldsB + (((g * kNChunk + kc) * 32 + lane) * 16));
#pragma unroll
      for (int g = 0; g < 4; ++g)
        acc[g] = __builtin_amdgcn_wmma_f32_16x16x32_f16(
            false, a, false, bfr[g], (short)0, acc[g], false, false);
    }

    // ---- K-phase 2: recurrence (h tile, chunks 8..23) ----
    const _Float16* ha = hprev + arow * kHid;
#pragma unroll
    for (int kc = 0; kc < kHChunk; ++kc) {
      v16h a = load_frag_g(ha + kc * 32, half8);
      v16h bfr[4];
#pragma unroll
      for (int g = 0; g < 4; ++g)
        bfr[g] = load_frag_lds(ldsB + (((g * kNChunk + kXChunk + kc) * 32 + lane) * 16));
#pragma unroll
      for (int g = 0; g < 4; ++g)
        acc[g] = __builtin_amdgcn_wmma_f32_16x16x32_f16(
            false, a, false, bfr[g], (short)0, acc[g], false, false);
    }

    // gates + state update; C/D layout: VGPR r, lane l -> (M = r + 8*(l>=16), N = l&15)
#pragma unroll
    for (int r = 0; r < 8; ++r) {
      int crow = m0 + r + half8;
      float iv = sigmoidf_(acc[0][r] + bias[0]);
      float fv = sigmoidf_(acc[1][r] + bias[1]);
      float gv = tanhf_(acc[2][r] + bias[2]);
      float ov = sigmoidf_(acc[3][r] + bias[3]);
      float cn = fv * creg[r] + iv * gv;
      creg[r]  = cn;
      float hv = ov * tanhf_(cn);
      out[((size_t)t * kBatch + crow) * kHid + jcol] = hv;        // hidden_seq
      hnext[crow * kHid + jcol] = (_Float16)hv;                   // next-step h
      if (t == kSeq - 1) {
        const size_t HS = (size_t)kSeq * kBatch * kHid;
        out[HS + (size_t)crow * kHid + jcol] = hv;                // h_t
        out[HS + (size_t)kBatch * kHid + (size_t)crow * kHid + jcol] = cn;  // c_t
      }
    }

    // device-scope split barrier between steps (h double-buffered)
    if (t + 1 < kSeq) {
      __threadfence();
      __syncthreads();
      if (tid == 0) {
        __hip_atomic_fetch_add(bar, 1, __ATOMIC_ACQ_REL, __HIP_MEMORY_SCOPE_AGENT);
        const int target = (t + 1) * (int)gridDim.x;
        while (__hip_atomic_load(bar, __ATOMIC_ACQUIRE, __HIP_MEMORY_SCOPE_AGENT) < target) {
          __builtin_amdgcn_s_sleep(1);
        }
      }
      __syncthreads();
      __threadfence();
    }
  }
}

// ---------------------------------------------------------------------------
extern "C" void kernel_launch(void* const* d_in, const int* in_sizes, int n_in,
                              void* d_out, int out_size, void* d_ws, size_t ws_size,
                              hipStream_t stream) {
  const float* x   = (const float*)d_in[0];
  const float* w_i = (const float*)d_in[1];
  const float* u_i = (const float*)d_in[2];
  const float* b_i = (const float*)d_in[3];
  const float* w_f = (const float*)d_in[4];
  const float* u_f = (const float*)d_in[5];
  const float* b_f = (const float*)d_in[6];
  const float* w_c = (const float*)d_in[7];
  const float* u_c = (const float*)d_in[8];
  const float* b_c = (const float*)d_in[9];
  const float* w_o = (const float*)d_in[10];
  const float* u_o = (const float*)d_in[11];
  const float* b_o = (const float*)d_in[12];
  float* out = (float*)d_out;

  // workspace layout
  _Float16* x16 = (_Float16*)d_ws;                                   // 16 MB
  _Float16* h16 = (_Float16*)((char*)d_ws + (size_t)kBatch * kSeq * kIn * 2);  // 64 KB
  int* bar      = (int*)((char*)h16 + (size_t)2 * kBatch * kHid * 2);

  (void)hipFuncSetAttribute((const void*)lstm_persistent_kernel,
                            hipFuncAttributeMaxDynamicSharedMemorySize, kLdsBytes);

  const int nx = kBatch * kSeq * kIn;
  pack_x_kernel<<<nx / 256, 256, 0, stream>>>(x, x16, nx);
  init_state_kernel<<<(2 * kBatch * kHid + 255) / 256, 256, 0, stream>>>(h16, bar);
  lstm_persistent_kernel<<<kNWG, kThreads, kLdsBytes, stream>>>(
      w_i, u_i, b_i, w_f, u_f, b_f, w_c, u_c, b_c, w_o, u_o, b_o,
      x16, h16, bar, out);
}
